// Attention_14156212208163
// MI455X (gfx1250) — matrix operand order repaired and checked by reading, not compiled, not run on hardware
//
#include <hip/hip_runtime.h>
#include <stdint.h>

#define BB   2
#define SS   2048
#define HH   16
#define DD   64
#define HID  1024
#define NEGINF (-10000.0f)

typedef __attribute__((ext_vector_type(16))) __bf16 v16bf;
typedef __attribute__((ext_vector_type(8)))  float  v8f;

union FragAB { v16bf v; uint4 q[2]; };

#define WMMA_BF16(A, B, C) \
  __builtin_amdgcn_wmma_f32_16x16x32_bf16(false, (A), false, (B), (short)0, (C), false, false)

__device__ __forceinline__ unsigned short f2bf(float f) {
  union { float f; uint32_t u; } a; a.f = f;
  uint32_t u = a.u;
  uint32_t r = u + 0x7fffu + ((u >> 16) & 1u);
  return (unsigned short)(r >> 16);
}

// ---------------- conversion kernels ----------------

__global__ void cvt_x_bf16(const float* __restrict__ x,
                           unsigned short* __restrict__ xbf, int n) {
  int i = blockIdx.x * 256 + threadIdx.x;
  if (i < n) xbf[i] = f2bf(x[i]);
}

// W is [K=1024][N=1024] row-major (x @ W). Store Wt[n][k] bf16 so WMMA
// B-fragments (lane = column n, contiguous k) are 16B vector loads.
__global__ void cvt_w_transpose(const float* __restrict__ W,
                                unsigned short* __restrict__ Wt) {
  int i = blockIdx.x * 256 + threadIdx.x;   // 1M threads
  int k = i >> 10, n = i & 1023;
  Wt[(size_t)n * HID + k] = f2bf(W[i]);
}

// ---------------- projection GEMM ----------------
// out[m,n] = sum_k xbf[m,k] * W[k,n] + bias[n], then *scale.
// M = B*S = 4096, N = K = 1024. One wave computes a 16(M) x 64(N) tile.
// Explicit double-buffering: step k+1's fragments are loaded into a second
// live set before step k's WMMAs consume the first, forcing partial
// s_wait_loadcnt instead of wait-0 serialization.
// vmode 0: store [B,H,S,D] bf16 (Q with scale=0.125, K with scale=1)
// vmode 1: store V transposed [B,H,D,S] bf16 (vectorized store over s)

#define PROJ_LOAD_STEP(A, Bv, KK)                                        \
  do {                                                                   \
    (A).q[0] = *(const uint4*)(xrow + (KK) + kbA);                       \
    (A).q[1] = *(const uint4*)(xrow + (KK) + kbA + 16);                  \
    _Pragma("unroll")                                                    \
    for (int nt = 0; nt < 4; nt++) {                                     \
      const unsigned short* wrow = wrow0 + (size_t)nt * 16 * HID;        \
      (Bv)[nt].q[0] = *(const uint4*)(wrow + (KK) + kbB);                \
      (Bv)[nt].q[1] = *(const uint4*)(wrow + (KK) + kbB + 16);           \
    }                                                                    \
  } while (0)

__global__ void proj64(const unsigned short* __restrict__ Xbf,
                       const unsigned short* __restrict__ Wt,
                       const float* __restrict__ bias,
                       unsigned short* __restrict__ Out,
                       float scale, int vmode) {
  int lane = threadIdx.x & 31, wave = threadIdx.x >> 5;
  int tile = blockIdx.x * 8 + wave;          // 4096 tiles total
  int mtile = tile >> 4;                     // 16 n64-tiles
  int ntile64 = tile & 15;
  int half  = lane >> 4, col = lane & 15, off8 = half * 8;

  int m  = mtile * 16 + col;                 // A row for this lane
  int n0 = ntile64 * 64 + col;               // B column (nt=0) for this lane
  const unsigned short* xrow  = Xbf + (size_t)m * HID;
  const unsigned short* wrow0 = Wt  + (size_t)n0 * HID;
  int kbA = half * 8;                        // A: e0..7 -> kbA+e ; e8..15 -> kbA+16+(e-8)
  int kbB = half * 8;                        // B: same k order as A

  v8f acc[4];
#pragma unroll
  for (int nt = 0; nt < 4; nt++) acc[nt] = (v8f){0.f,0.f,0.f,0.f,0.f,0.f,0.f,0.f};

  FragAB aC, bC[4], aN, bN[4];
  PROJ_LOAD_STEP(aC, bC, 0);                 // prologue

#pragma unroll 2
  for (int kk = 0; kk < HID - 32; kk += 32) {
    PROJ_LOAD_STEP(aN, bN, kk + 32);         // next step's loads in flight
#pragma unroll
    for (int nt = 0; nt < 4; nt++)
      acc[nt] = WMMA_BF16(aC.v, bC[nt].v, acc[nt]);
    aC = aN;
#pragma unroll
    for (int nt = 0; nt < 4; nt++) bC[nt] = bN[nt];
  }
#pragma unroll
  for (int nt = 0; nt < 4; nt++)             // peeled last step
    acc[nt] = WMMA_BF16(aC.v, bC[nt].v, acc[nt]);

#pragma unroll
  for (int nt = 0; nt < 4; nt++) {
    int n = n0 + nt * 16;
    float bn = bias[n];
    int h = n >> 6, d = n & 63;
    if (vmode == 0) {
#pragma unroll
      for (int r = 0; r < 8; r++) {
        int t = mtile * 16 + r + off8;       // token index over B*S
        int bb = t >> 11, s = t & (SS - 1);
        float v = (acc[nt][r] + bn) * scale;
        Out[(((size_t)(bb * HH + h)) * SS + s) * DD + d] = f2bf(v);
      }
    } else {
      // 16 rows of a tile stay within one batch (S % 16 == 0)
      int t0 = mtile * 16 + off8;
      int bb = t0 >> 11, s0 = t0 & (SS - 1);
      union { unsigned short h16[8]; uint4 q; } pk;
#pragma unroll
      for (int r = 0; r < 8; r++) pk.h16[r] = f2bf((acc[nt][r] + bn) * scale);
      *(uint4*)(Out + (((size_t)(bb * HH + h)) * DD + d) * SS + s0) = pk.q;
    }
  }
}

// ---------------- flash attention ----------------
// One wave per 16-query tile of one (b,h). Stream keys 32 at a time.
// V fragments for the P*V step are loaded at the TOP of each chunk so the
// global loads stay in flight across the score WMMAs and the softmax VALU
// block; only the LDS P-transpose read is gated by s_wait_dscnt.
__global__ void attn_flash(const unsigned short* __restrict__ Qbf,  // [B,H,S,D]
                           const unsigned short* __restrict__ Kbf,  // [B,H,S,D]
                           const unsigned short* __restrict__ Vt,   // [B,H,D,S]
                           const int* __restrict__ mask,            // [B,S]
                           float* __restrict__ Out) {               // [B,S,HID]
  __shared__ __align__(16) unsigned short ldsP[8][16][40];  // per-wave P tile

  int lane = threadIdx.x & 31, wave = threadIdx.x >> 5;
  int tile = blockIdx.x * 8 + wave;
  int qt = tile & (SS / 16 - 1);
  int bh = tile >> 7;                       // S/16 = 128
  int b = bh >> 4, h = bh & (HH - 1);
  int half = lane >> 4, col = lane & 15, off8 = half * 8;
  int qrow = qt * 16 + col;

  // Q row A-fragments, d = 0..63 (two 16x32 frags), resident for whole loop
  const unsigned short* qptr = Qbf + ((size_t)bh * SS + qrow) * DD;
  FragAB qa0, qa1;
  qa0.q[0] = *(const uint4*)(qptr + half * 16);
  qa0.q[1] = *(const uint4*)(qptr + half * 16 + 8);
  qa1.q[0] = *(const uint4*)(qptr + 32 + half * 16);
  qa1.q[1] = *(const uint4*)(qptr + 32 + half * 16 + 8);

  float mrow[8], lrow[8];
  v8f acc[4];
#pragma unroll
  for (int r = 0; r < 8; r++) { mrow[r] = -1e30f; lrow[r] = 0.f; }
#pragma unroll
  for (int dt = 0; dt < 4; dt++) acc[dt] = (v8f){0.f,0.f,0.f,0.f,0.f,0.f,0.f,0.f};

  const int* mrowp = mask + b * SS;
  const unsigned short* kbase_all = Kbf + ((size_t)bh * SS + col) * DD;
  const unsigned short* vbase_all = Vt + ((size_t)bh * DD + col) * SS;

  for (int kc = 0; kc < SS; kc += 32) {
    // prefetch next chunk's K/V lines (global_prefetch_b8) behind this chunk
    if (kc + 32 < SS) {
      __builtin_prefetch(kbase_all + (size_t)(kc + 32) * DD, 0, 3);
      __builtin_prefetch(vbase_all + kc + 32, 0, 3);
    }

    // ---- V B-fragments for this chunk, issued early (consumed at the end)
    FragAB vb[4];
#pragma unroll
    for (int dt = 0; dt < 4; dt++) {
      const unsigned short* vptr =
          Vt + ((size_t)bh * DD + dt * 16 + col) * SS + kc + half * 16;
      vb[dt].q[0] = *(const uint4*)(vptr);
      vb[dt].q[1] = *(const uint4*)(vptr + 8);
    }

    // ---- scores: two 16x16 C tiles over keys [kc..kc+15] and [kc+16..kc+31]
    const unsigned short* kb0 = kbase_all + (size_t)kc * DD;
    const unsigned short* kb1 = kb0 + 16 * DD;
    FragAB kb00, kb01, kb10, kb11;
    kb00.q[0] = *(const uint4*)(kb0 + half * 16);
    kb00.q[1] = *(const uint4*)(kb0 + half * 16 + 8);
    kb01.q[0] = *(const uint4*)(kb0 + 32 + half * 16);
    kb01.q[1] = *(const uint4*)(kb0 + 32 + half * 16 + 8);
    kb10.q[0] = *(const uint4*)(kb1 + half * 16);
    kb10.q[1] = *(const uint4*)(kb1 + half * 16 + 8);
    kb11.q[0] = *(const uint4*)(kb1 + 32 + half * 16);
    kb11.q[1] = *(const uint4*)(kb1 + 32 + half * 16 + 8);

    v8f c0 = (v8f){0.f,0.f,0.f,0.f,0.f,0.f,0.f,0.f};
    v8f c1 = (v8f){0.f,0.f,0.f,0.f,0.f,0.f,0.f,0.f};
    c0 = WMMA_BF16(qa0.v, kb00.v, c0);
    c0 = WMMA_BF16(qa1.v, kb01.v, c0);
    c1 = WMMA_BF16(qa0.v, kb10.v, c1);
    c1 = WMMA_BF16(qa1.v, kb11.v, c1);

    // additive padding mask (per key column, same for all rows of this lane)
    float add0 = (1.0f - (float)mrowp[kc + col]) * NEGINF;
    float add1 = (1.0f - (float)mrowp[kc + 16 + col]) * NEGINF;

    // ---- online softmax update; C layout: lane = col, element r = row r+off8
#pragma unroll
    for (int r = 0; r < 8; r++) {
      float s0 = c0[r] + add0;
      float s1 = c1[r] + add1;
      float t = fmaxf(s0, s1);
      t = fmaxf(t, __shfl_xor(t, 8, 32));
      t = fmaxf(t, __shfl_xor(t, 4, 32));
      t = fmaxf(t, __shfl_xor(t, 2, 32));
      t = fmaxf(t, __shfl_xor(t, 1, 32));
      float nm = fmaxf(mrow[r], t);
      float alpha = __expf(mrow[r] - nm);
      float p0 = __expf(s0 - nm);
      float p1 = __expf(s1 - nm);
      float ps = p0 + p1;
      ps += __shfl_xor(ps, 8, 32);
      ps += __shfl_xor(ps, 4, 32);
      ps += __shfl_xor(ps, 2, 32);
      ps += __shfl_xor(ps, 1, 32);
      lrow[r] = lrow[r] * alpha + ps;
      mrow[r] = nm;
      acc[0][r] *= alpha; acc[1][r] *= alpha;
      acc[2][r] *= alpha; acc[3][r] *= alpha;
      ldsP[wave][r + off8][col]      = f2bf(p0);
      ldsP[wave][r + off8][16 + col] = f2bf(p1);
    }

    // DS ops are in-order per wave; just stop compiler reordering.
    asm volatile("" ::: "memory");
    asm volatile("s_wait_dscnt 0" ::: "memory");

    // ---- P as 16x32 A-fragment
    FragAB pA;
    pA.q[0] = *(const uint4*)&ldsP[wave][col][half * 16];
    pA.q[1] = *(const uint4*)&ldsP[wave][col][half * 16 + 8];
    asm volatile("" ::: "memory");

    // ---- ctx += P (16x32) x V (32x64) using the preloaded V fragments
#pragma unroll
    for (int dt = 0; dt < 4; dt++)
      acc[dt] = WMMA_BF16(pA.v, vb[dt].v, acc[dt]);
  }

  // ---- epilogue: normalize and store f32 [B,S,HID]
#pragma unroll
  for (int dt = 0; dt < 4; dt++) {
#pragma unroll
    for (int r = 0; r < 8; r++) {
      int s = qt * 16 + r + off8;
      Out[((size_t)(b * SS + s)) * HID + h * DD + dt * 16 + col] =
          acc[dt][r] / lrow[r];
    }
  }
}

// ---------------- launch ----------------

extern "C" void kernel_launch(void* const* d_in, const int* in_sizes, int n_in,
                              void* d_out, int out_size, void* d_ws, size_t ws_size,
                              hipStream_t stream) {
  const float* x    = (const float*)d_in[0];
  const int*   mask = (const int*)d_in[1];
  const float* Wq   = (const float*)d_in[2];
  const float* bq   = (const float*)d_in[3];
  const float* Wk   = (const float*)d_in[4];
  const float* bk   = (const float*)d_in[5];
  const float* Wv   = (const float*)d_in[6];
  const float* bv   = (const float*)d_in[7];
  float* out = (float*)d_out;

  char* ws = (char*)d_ws;
  size_t o = 0;
  auto alloc = [&](size_t bytes) -> void* {
    void* p = ws + o;
    o += (bytes + 255) & ~(size_t)255;
    return p;
  };
  const size_t nX = (size_t)BB * SS * HID;      // 4,194,304
  const size_t nW = (size_t)HID * HID;          // 1,048,576
  unsigned short* xbf = (unsigned short*)alloc(nX * 2);
  unsigned short* Wqt = (unsigned short*)alloc(nW * 2);
  unsigned short* Wkt = (unsigned short*)alloc(nW * 2);
  unsigned short* Wvt = (unsigned short*)alloc(nW * 2);
  unsigned short* Qb  = (unsigned short*)alloc(nX * 2);  // [B,H,S,D]
  unsigned short* Kb  = (unsigned short*)alloc(nX * 2);  // [B,H,S,D]
  unsigned short* Vtr = (unsigned short*)alloc(nX * 2);  // [B,H,D,S]

  cvt_x_bf16<<<(int)(nX / 256), 256, 0, stream>>>(x, xbf, (int)nX);
  cvt_w_transpose<<<(int)(nW / 256), 256, 0, stream>>>(Wq, Wqt);
  cvt_w_transpose<<<(int)(nW / 256), 256, 0, stream>>>(Wk, Wkt);
  cvt_w_transpose<<<(int)(nW / 256), 256, 0, stream>>>(Wv, Wvt);

  // wave-tiles = (4096/16)*(1024/64) = 4096 ; 8 waves/block
  proj64<<<512, 256, 0, stream>>>(xbf, Wqt, bq, Qb, 0.125f, 0); // Q pre-scaled 1/sqrt(D)
  proj64<<<512, 256, 0, stream>>>(xbf, Wkt, bk, Kb, 1.0f, 0);
  proj64<<<512, 256, 0, stream>>>(xbf, Wvt, bv, Vtr, 1.0f, 1);  // V transposed

  // query tiles = B*H*(S/16) = 4096 ; 8 waves/block
  attn_flash<<<512, 256, 0, stream>>>(Qb, Kb, Vtr, mask, out);
}
